// LIFPolicy_61177514164829
// MI455X (gfx1250) — matrix operand order repaired and checked by reading, not compiled, not run on hardware
//
#include <hip/hip_runtime.h>
#include <hip/hip_bf16.h>

#define BETA_C   0.9f
#define VTH_C    1.0f
#define NB       256    // batch
#define TSEQ     512    // sequence length
#define OBS      64     // obs dim
#define NN       1024   // neurons
#define ADIM     64     // action dim
#define LDP      1048   // padded LDS row stride (halves): 2096 B, 16B-aligned, bank-spread

typedef _Float16 v8h  __attribute__((ext_vector_type(8)));
typedef _Float16 v16h __attribute__((ext_vector_type(16)));
typedef float    v8f  __attribute__((ext_vector_type(8)));

// ---------------------------------------------------------------------------
// WMMA fragment loaders (CDNA5 ISA 7.12.2 layouts, wave32)
// ---------------------------------------------------------------------------

// A-matrix 16x32 f16 fragment from row-major f16 matrix (global).
// lanes 0-15: M=lane, K = k0+{0..7} (v[0..7]) and k0+{16..23} (v[8..15]);
// lanes 16-31: M=lane-16, K shifted by +8 / +24.
__device__ __forceinline__ v16h load_fragA_h(const _Float16* p, int row0, int k0, int ld) {
  const int lane = (int)(threadIdx.x & 31u);
  const _Float16* q = p + (size_t)(row0 + (lane & 15)) * ld + k0 + ((lane >> 4) << 3);
  v8h lo = *(const v8h*)(q);
  v8h hi = *(const v8h*)(q + 16);
  v16h r;
#pragma unroll
  for (int i = 0; i < 8; ++i) { r[i] = lo[i]; r[8 + i] = hi[i]; }
  return r;
}

// Same A layout sourced from the LDS spike panel (row stride LDP halves).
__device__ __forceinline__ v16h load_fragA_lds(const _Float16* sA, int k0) {
  const int lane = (int)(threadIdx.x & 31u);
  const _Float16* q = sA + (lane & 15) * LDP + k0 + ((lane >> 4) << 3);
  v8h lo = *(const v8h*)(q);
  v8h hi = *(const v8h*)(q + 16);
  v16h r;
#pragma unroll
  for (int i = 0; i < 8; ++i) { r[i] = lo[i]; r[8 + i] = hi[i]; }
  return r;
}

// Same A layout sourced from an f32 row-major matrix (convert on the fly).
__device__ __forceinline__ v16h load_fragA_f32(const float* p, int row0, int k0, int ld) {
  const int lane = (int)(threadIdx.x & 31u);
  const float* q = p + (size_t)(row0 + (lane & 15)) * ld + k0 + ((lane >> 4) << 3);
  v16h r;
#pragma unroll
  for (int i = 0; i < 8; ++i) { r[i] = (_Float16)q[i]; r[8 + i] = (_Float16)q[16 + i]; }
  return r;
}

// B-matrix 32x16 f16 fragment where B[k][n] = W[n][k], W row-major, stride ld.
// lanes 0-15: n=lane, K = k0+{0..7} (v[0..7]) and k0+{16..23} (v[8..15]);
// lanes 16-31: n=lane-16, K shifted by +8 / +24 (matches A fragment).
__device__ __forceinline__ v16h load_fragB_h(const _Float16* p, int col0, int k0, int ld) {
  const int lane = (int)(threadIdx.x & 31u);
  const _Float16* q = p + (size_t)(col0 + (lane & 15)) * ld + k0 + ((lane >> 4) << 3);
  v8h lo = *(const v8h*)(q);
  v8h hi = *(const v8h*)(q + 16);
  v16h r;
#pragma unroll
  for (int i = 0; i < 8; ++i) { r[i] = lo[i]; r[8 + i] = hi[i]; }
  return r;
}

// ---------------------------------------------------------------------------
// Device-wide generation barrier (one per timestep)
// ---------------------------------------------------------------------------
__device__ __forceinline__ void grid_barrier(unsigned* bar, unsigned nblk) {
  __syncthreads();
  if (threadIdx.x == 0) {
    unsigned* cnt = bar;
    unsigned* gen = bar + 1;
    unsigned g = __atomic_load_n(gen, __ATOMIC_RELAXED);
    if (__atomic_add_fetch(cnt, 1u, __ATOMIC_ACQ_REL) == nblk) {
      __atomic_store_n(cnt, 0u, __ATOMIC_RELAXED);
      __atomic_store_n(gen, g + 1u, __ATOMIC_RELEASE);
    } else {
      while (__atomic_load_n(gen, __ATOMIC_ACQUIRE) == g) {
        __builtin_amdgcn_s_sleep(2);
      }
    }
  }
  __syncthreads();
}

// ---------------------------------------------------------------------------
// Prep: f32 -> f16 weight conversion, zero spike buffers + barrier state
// ---------------------------------------------------------------------------
__global__ void lif_prep(const float* __restrict__ Wrec, const float* __restrict__ Win,
                         const float* __restrict__ Wout,
                         _Float16* __restrict__ Wrec16, _Float16* __restrict__ Win16,
                         _Float16* __restrict__ Wout16,
                         _Float16* __restrict__ s0, _Float16* __restrict__ s1,
                         unsigned* __restrict__ bar) {
  size_t i = (size_t)blockIdx.x * blockDim.x + threadIdx.x;
  size_t stride = (size_t)gridDim.x * blockDim.x;
  for (size_t j = i; j < (size_t)NN * NN; j += stride) Wrec16[j] = (_Float16)Wrec[j];
  for (size_t j = i; j < (size_t)NN * OBS; j += stride) Win16[j] = (_Float16)Win[j];
  for (size_t j = i; j < (size_t)ADIM * NN; j += stride) Wout16[j] = (_Float16)Wout[j];
  for (size_t j = i; j < (size_t)NB * NN; j += stride) { s0[j] = (_Float16)0.0f; s1[j] = (_Float16)0.0f; }
  if (i < 2) bar[i] = 0u;
}

// ---------------------------------------------------------------------------
// Persistent LIF recurrence kernel.
// 32 blocks x 512 threads = 512 wave32 waves. Wave w owns batch-tile bt=w>>5
// and two adjacent neuron tiles nt = 2*(w&31), 2*(w&31)+1. All waves of a
// block share one batch tile, whose spike panel is staged in LDS each step.
// Membrane potential v lives in VGPR accumulators across all 512 steps.
// ---------------------------------------------------------------------------
__global__ __launch_bounds__(512) void lif_persistent(
    const float* __restrict__ inputs,        // (NB, TSEQ, OBS) f32
    const _Float16* __restrict__ Wrec,       // (NN, NN) f16
    const _Float16* __restrict__ Win,        // (NN, OBS) f16
    const _Float16* __restrict__ Wout,       // (ADIM, NN) f16
    _Float16* __restrict__ s0buf, _Float16* __restrict__ s1buf,  // spikes ping-pong
    _Float16* __restrict__ v0buf, _Float16* __restrict__ v1buf,  // v f16 (double buffer)
    float* __restrict__ out,                 // (NB, TSEQ, ADIM) f32
    unsigned* __restrict__ bar) {
  __shared__ _Float16 sA[16 * LDP];          // one 16-row spike panel (padded)

  const int lane = (int)(threadIdx.x & 31u);
  const int wave = (int)((blockIdx.x * blockDim.x + threadIdx.x) >> 5);  // 0..511
  const int m0   = (lane >> 4) << 3;   // C/D layout: row offset 0 or 8
  const int col  = lane & 15;          // C/D layout: column = lane%16

  const int bt = wave >> 5;            // 0..15 (uniform within a block)
  const int b0 = bt * 16;
  const int n0 = ((wave & 31) * 2) * 16;
  const int n1 = n0 + 16;

  v8f vreg[2], sreg[2];
#pragma unroll
  for (int u = 0; u < 2; ++u)
#pragma unroll
    for (int i = 0; i < 8; ++i) { vreg[u][i] = 0.0f; sreg[u][i] = 0.0f; }

  // Output-head tiles: one head wave per 8 waves (64 head tiles, 2 per block).
  const bool doHead = ((wave & 7) == 0);
  const int  tileB  = wave >> 3;            // 0..63
  const int  bB0    = (tileB >> 2) * 16;    // batch tile origin
  const int  aB0    = (tileB & 3) * 16;     // action tile origin

  for (int t = 0; t < TSEQ; ++t) {
    const _Float16* sprev = (t & 1) ? s1buf : s0buf;
    _Float16* snext = (t & 1) ? s0buf : s1buf;
    _Float16* vcur  = (t & 1) ? v1buf : v0buf;

    // ---- Stage this block's 16-row spike panel into LDS (32 KB, 2048 x 16B).
    {
      const _Float16* src = sprev + (size_t)b0 * NN;
#pragma unroll
      for (int c = (int)threadIdx.x; c < 2048; c += 512) {
        const int row = c >> 7;           // 128 16B-chunks per 1024-half row
        const int off = (c & 127) * 8;
        *(v8h*)(&sA[row * LDP + off]) = *(const v8h*)(src + (size_t)row * NN + off);
      }
    }
    __syncthreads();

    // ---- Phase A: v_t = beta*v + s@Wrec^T + x_t@Win^T - vth*s ; s_t = v_t > vth
    v8f acc0, acc1;
#pragma unroll
    for (int i = 0; i < 8; ++i) {
      acc0[i] = BETA_C * vreg[0][i] - VTH_C * sreg[0][i];
      acc1[i] = BETA_C * vreg[1][i] - VTH_C * sreg[1][i];
    }

#pragma unroll 4
    for (int kk = 0; kk < NN; kk += 32) {
      v16h a  = load_fragA_lds(sA, kk);                 // shared by both n-tiles
      v16h w0 = load_fragB_h(Wrec, n0, kk, NN);
      v16h w1 = load_fragB_h(Wrec, n1, kk, NN);
      acc0 = __builtin_amdgcn_wmma_f32_16x16x32_f16(false, a, false, w0, (short)0, acc0,
                                                    false, false);
      acc1 = __builtin_amdgcn_wmma_f32_16x16x32_f16(false, a, false, w1, (short)0, acc1,
                                                    false, false);
    }
#pragma unroll
    for (int oo = 0; oo < OBS; oo += 32) {
      v16h a  = load_fragA_f32(inputs + (size_t)t * OBS, b0, oo, TSEQ * OBS);
      v16h w0 = load_fragB_h(Win, n0, oo, OBS);
      v16h w1 = load_fragB_h(Win, n1, oo, OBS);
      acc0 = __builtin_amdgcn_wmma_f32_16x16x32_f16(false, a, false, w0, (short)0, acc0,
                                                    false, false);
      acc1 = __builtin_amdgcn_wmma_f32_16x16x32_f16(false, a, false, w1, (short)0, acc1,
                                                    false, false);
    }

    vreg[0] = acc0; vreg[1] = acc1;
#pragma unroll
    for (int i = 0; i < 8; ++i) {
      sreg[0][i] = (acc0[i] > VTH_C) ? 1.0f : 0.0f;
      sreg[1][i] = (acc1[i] > VTH_C) ? 1.0f : 0.0f;
    }

    // Scatter C-layout tiles to row-major f16 buffers (v copy + next spikes).
#pragma unroll
    for (int r = 0; r < 8; ++r) {
      const size_t row = (size_t)(b0 + m0 + r) * NN;
      vcur[row + n0 + col]  = (_Float16)acc0[r];
      snext[row + n0 + col] = (_Float16)sreg[0][r];
      vcur[row + n1 + col]  = (_Float16)acc1[r];
      snext[row + n1 + col] = (_Float16)sreg[1][r];
    }

    // ---- Overlapped action head for step t-1: tanh(v_{t-1} @ Wout^T)
    if (doHead && t > 0) {
      const _Float16* vprev = ((t - 1) & 1) ? v1buf : v0buf;
      v8f acc;
#pragma unroll
      for (int i = 0; i < 8; ++i) acc[i] = 0.0f;
#pragma unroll 4
      for (int kk = 0; kk < NN; kk += 32) {
        v16h a = load_fragA_h(vprev, bB0, kk, NN);
        v16h b = load_fragB_h(Wout, aB0, kk, NN);
        acc = __builtin_amdgcn_wmma_f32_16x16x32_f16(false, a, false, b, (short)0, acc,
                                                     false, false);
      }
#pragma unroll
      for (int r = 0; r < 8; ++r) {
        const size_t o = (size_t)(bB0 + m0 + r) * (TSEQ * ADIM) +
                         (size_t)(t - 1) * ADIM + (size_t)(aB0 + col);
        out[o] = tanhf(acc[r]);
      }
    }

    __threadfence();
    grid_barrier(bar, gridDim.x);
  }

  // Final head for t = TSEQ-1 (published before the last barrier).
  if (doHead) {
    const _Float16* vprev = ((TSEQ - 1) & 1) ? v1buf : v0buf;
    v8f acc;
#pragma unroll
    for (int i = 0; i < 8; ++i) acc[i] = 0.0f;
#pragma unroll 4
    for (int kk = 0; kk < NN; kk += 32) {
      v16h a = load_fragA_h(vprev, bB0, kk, NN);
      v16h b = load_fragB_h(Wout, aB0, kk, NN);
      acc = __builtin_amdgcn_wmma_f32_16x16x32_f16(false, a, false, b, (short)0, acc,
                                                   false, false);
    }
#pragma unroll
    for (int r = 0; r < 8; ++r) {
      const size_t o = (size_t)(bB0 + m0 + r) * (TSEQ * ADIM) +
                       (size_t)(TSEQ - 1) * ADIM + (size_t)(aB0 + col);
      out[o] = tanhf(acc[r]);
    }
  }
}

// ---------------------------------------------------------------------------
// Launch
// ---------------------------------------------------------------------------
extern "C" void kernel_launch(void* const* d_in, const int* in_sizes, int n_in,
                              void* d_out, int out_size, void* d_ws, size_t ws_size,
                              hipStream_t stream) {
  (void)in_sizes; (void)n_in; (void)out_size; (void)ws_size;
  const float* inputs = (const float*)d_in[0];   // (256, 512, 64)
  const float* Wrec   = (const float*)d_in[1];   // (1024, 1024)
  const float* Win    = (const float*)d_in[2];   // (1024, 64)
  const float* Wout   = (const float*)d_in[3];   // (64, 1024)
  float* out = (float*)d_out;

  // Workspace layout (f16 elements): weights, spike ping-pong, v double buffer.
  _Float16* w       = (_Float16*)d_ws;
  _Float16* Wrec16  = w;
  _Float16* Win16   = Wrec16 + (size_t)NN * NN;
  _Float16* Wout16  = Win16 + (size_t)NN * OBS;
  _Float16* s0      = Wout16 + (size_t)ADIM * NN;
  _Float16* s1      = s0 + (size_t)NB * NN;
  _Float16* v0      = s1 + (size_t)NB * NN;
  _Float16* v1      = v0 + (size_t)NB * NN;
  unsigned* bar     = (unsigned*)(v1 + (size_t)NB * NN);  // 2 words, 8B aligned

  hipLaunchKernelGGL(lif_prep, dim3(512), dim3(256), 0, stream,
                     Wrec, Win, Wout, Wrec16, Win16, Wout16, s0, s1, bar);

  hipLaunchKernelGGL(lif_persistent, dim3(32), dim3(512), 0, stream,
                     inputs, Wrec16, Win16, Wout16, s0, s1, v0, v1, out, bar);
}
